// AttentionLayer_38173669327473
// MI455X (gfx1250) — matrix operand order repaired and checked by reading, compiled, not run on hardware
//
#include <hip/hip_runtime.h>

// ---------------------------------------------------------------------------
// Single-head attention: Q=xWq+bq, K=xWk+bk, V=xWv+bv, O = softmax(QK^T/32) V
// B=4, S=2048, D=H=1024.
// GEMMs: v_wmma_f32_16x16x32_bf16, async global->LDS DMA, double-buffered LDS.
// ---------------------------------------------------------------------------

#define ATT_B 4
#define ATT_S 2048
#define ATT_D 1024
#define ATT_H 1024

typedef __attribute__((ext_vector_type(16))) __bf16 v16bf;
typedef __attribute__((ext_vector_type(8)))  float  v8f;

union FragB16 { v16bf v; uint4 q[2]; };

__device__ __forceinline__ unsigned short f2bf(float f) {
    unsigned int u = __float_as_uint(f);
    unsigned int r = (u + 0x7FFFu + ((u >> 16) & 1u)) >> 16;   // RNE
    return (unsigned short)r;
}
__device__ __forceinline__ float bf2f(unsigned short h) {
    return __uint_as_float(((unsigned int)h) << 16);
}

// CDNA5 async DMA: 16B global -> LDS, tracked by ASYNCcnt (no VGPR data path).
__device__ __forceinline__ void async_copy16(const unsigned short* g,
                                             const unsigned short* l) {
    unsigned ldsOff = (unsigned)(uintptr_t)l;              // low 32 bits = LDS offset
    unsigned long long ga = (unsigned long long)(uintptr_t)g;
    asm volatile("global_load_async_to_lds_b128 %0, %1, off"
                 :: "v"(ldsOff), "v"(ga) : "memory");
}

// ---------------------------------------------------------------------------
// fp32 -> bf16 elementwise convert (vector-4)
// ---------------------------------------------------------------------------
__global__ __launch_bounds__(256) void convert_bf16_v4(const float* __restrict__ in,
                                                       unsigned short* __restrict__ out,
                                                       long long n4) {
    long long t = (long long)blockIdx.x * blockDim.x + threadIdx.x;
    if (t >= n4) return;
    float4 f = ((const float4*)in)[t];
    union { unsigned short u[4]; uint2 v; } pk;
    pk.u[0] = f2bf(f.x); pk.u[1] = f2bf(f.y);
    pk.u[2] = f2bf(f.z); pk.u[3] = f2bf(f.w);
    ((uint2*)out)[t] = pk.v;
}

// ---------------------------------------------------------------------------
// W[k][n] fp32 -> Wt[n][k] bf16 (so WMMA B-operand is K-contiguous per column)
// ---------------------------------------------------------------------------
__global__ __launch_bounds__(256) void transpose_bf16(const float* __restrict__ W,
                                                      unsigned short* __restrict__ Wt,
                                                      int Kd, int Nd) {
    long long t = (long long)blockIdx.x * blockDim.x + threadIdx.x;
    if (t >= (long long)Kd * Nd) return;
    int n = (int)(t / Kd);
    int k = (int)(t - (long long)n * Kd);
    Wt[t] = f2bf(W[(long long)k * Nd + n]);
}

// ---------------------------------------------------------------------------
// Generic WMMA GEMM:  C = scale * (A[M,K] x Bt[N,K]^T) + bias
//   A, Bt bf16 row-major with leading dim K.
//   mode 0: C bf16 [M,N]          (Q, K projections; raw scores)
//   mode 1: C bf16 transposed V:  C[b*N*S + n*S + s], b = row/Sdim (V^T store)
//   mode 2: C fp32 [M,N]          (final output)
//   blockIdx.z batches via zStride{A,B,C} (element strides).
// Block tile 256x128, 8 waves (4x2), wave tile 64x64, K-chunk 32,
// double-buffered LDS fed by global_load_async_to_lds_b128.
// ---------------------------------------------------------------------------
#define BM 256
#define BN 128
#define TK 32
#define PK 40   // LDS row pitch in bf16 elems (32 + 8 pad, keeps 16B alignment)

__global__ __launch_bounds__(256) void wmma_gemm(const unsigned short* __restrict__ A,
                                                 const unsigned short* __restrict__ Bt,
                                                 const float* __restrict__ bias,
                                                 float scale,
                                                 void* __restrict__ Cout,
                                                 int M, int N, int K,
                                                 long long zStrideA, long long zStrideB,
                                                 long long zStrideC,
                                                 int mode, int Sdim) {
    __shared__ __align__(16) unsigned short sA[2][BM * PK];   // 2 x 20480 B
    __shared__ __align__(16) unsigned short sB[2][BN * PK];   // 2 x 10240 B

    A  += (long long)blockIdx.z * zStrideA;
    Bt += (long long)blockIdx.z * zStrideB;

    const int tid    = threadIdx.x;
    const int waveId = tid >> 5;
    const int lane   = tid & 31;
    const int l15    = lane & 15;
    const int half   = lane >> 4;
    const int wm     = waveId >> 1;   // 0..3  (64-row slot)
    const int wn     = waveId & 1;    // 0..1  (64-col slot)

    const int m0 = blockIdx.y * BM;
    const int n0 = blockIdx.x * BN;

    // Stage one 256x32 A tile (4 chunks/thread) + 128x32 B tile (2 chunks/thread):
    // 6 async 16B DMAs per lane per K-step.
    auto stage = [&](int k0, int bsel) {
#pragma unroll
        for (int i = 0; i < 4; ++i) {
            int c    = tid + i * 256;      // 0..1023
            int row  = c >> 2;             // 0..255
            int col8 = (c & 3) * 8;
            async_copy16(&A[(long long)(m0 + row) * K + k0 + col8],
                         &sA[bsel][row * PK + col8]);
        }
#pragma unroll
        for (int i = 0; i < 2; ++i) {
            int c    = tid + i * 256;      // 0..511
            int row  = c >> 2;             // 0..127
            int col8 = (c & 3) * 8;
            async_copy16(&Bt[(long long)(n0 + row) * K + k0 + col8],
                         &sB[bsel][row * PK + col8]);
        }
    };

    v8f acc[4][4];
    const v8f vzero = {0.f, 0.f, 0.f, 0.f, 0.f, 0.f, 0.f, 0.f};
#pragma unroll
    for (int mi = 0; mi < 4; ++mi)
#pragma unroll
        for (int ni = 0; ni < 4; ++ni) acc[mi][ni] = vzero;

    const int nT = K / TK;
    stage(0, 0);

    for (int t = 0; t < nT; ++t) {
        const int cur = t & 1;
        if (t + 1 < nT) {
            stage((t + 1) * TK, cur ^ 1);                 // DMA next tile
            asm volatile("s_wait_asynccnt 0x6" ::: "memory");  // cur tile landed
        } else {
            asm volatile("s_wait_asynccnt 0x0" ::: "memory");
        }
        __syncthreads();

        // A fragments: 16x32, lanes0-15 K{0-7,16-23}, lanes16-31 K{8-15,24-31}
        FragB16 aF[4];
#pragma unroll
        for (int mi = 0; mi < 4; ++mi) {
            const unsigned short* p =
                &sA[cur][(wm * 64 + mi * 16 + l15) * PK + half * 8];
            aF[mi].q[0] = *(const uint4*)p;
            aF[mi].q[1] = *(const uint4*)(p + 16);
        }
        // B fragments one at a time (keeps VGPR pressure down): column = lane,
        // lanes0-15 K{0-7,16-23}, lanes16-31 K{8-15,24-31}.
#pragma unroll
        for (int ni = 0; ni < 4; ++ni) {
            FragB16 bF;
            const unsigned short* p =
                &sB[cur][(wn * 64 + ni * 16 + l15) * PK + half * 8];
            bF.q[0] = *(const uint4*)p;
            bF.q[1] = *(const uint4*)(p + 16);
#pragma unroll
            for (int mi = 0; mi < 4; ++mi)
                acc[mi][ni] = __builtin_amdgcn_wmma_f32_16x16x32_bf16(
                    false, aF[mi].v, false, bF.v,
                    (short)0, acc[mi][ni], false, false);
        }
        __syncthreads();   // all waves done reading 'cur' before it is re-DMA'd
    }

    // ---- epilogue: C layout -> lanes0-15 N=lane M=r; lanes16-31 N=lane-16 M=8+r
#pragma unroll
    for (int mi = 0; mi < 4; ++mi) {
#pragma unroll
        for (int ni = 0; ni < 4; ++ni) {
            const int col  = n0 + wn * 64 + ni * 16 + l15;
            const int rowb = m0 + wm * 64 + mi * 16 + half * 8;
            const float bv = bias ? bias[col] : 0.0f;
            v8f c = acc[mi][ni];
            if (mode == 0) {
                unsigned short* Cb =
                    (unsigned short*)Cout + (long long)blockIdx.z * zStrideC;
#pragma unroll
                for (int r = 0; r < 8; ++r)
                    Cb[(long long)(rowb + r) * N + col] = f2bf(c[r] * scale + bv);
            } else if (mode == 1) {
                // V^T store: Vt[b][h][s]; 8 consecutive s -> one 16B store
                union { unsigned short u[8]; uint4 q; } pk;
#pragma unroll
                for (int r = 0; r < 8; ++r) pk.u[r] = f2bf(c[r] * scale + bv);
                int bb = rowb / Sdim;
                int sl = rowb - bb * Sdim;
                unsigned short* Cb = (unsigned short*)Cout;
                *(uint4*)&Cb[((long long)bb * N + col) * Sdim + sl] = pk.q;
            } else {
                float* Cf = (float*)Cout + (long long)blockIdx.z * zStrideC;
#pragma unroll
                for (int r = 0; r < 8; ++r)
                    Cf[(long long)(rowb + r) * N + col] = c[r] * scale;
            }
        }
    }
}

// ---------------------------------------------------------------------------
// Row softmax over bf16 scores, in place. One block per row of length L=2048.
// ---------------------------------------------------------------------------
__global__ __launch_bounds__(256) void softmax_rows(unsigned short* __restrict__ Sc, int L) {
    __shared__ float red[256];
    unsigned short* p = Sc + (long long)blockIdx.x * L;
    const int t = threadIdx.x;

    union { uint4 q; unsigned short u[8]; } pk;
    pk.q = *(const uint4*)(p + t * 8);
    float v[8];
    float mx = -3.4e38f;
#pragma unroll
    for (int i = 0; i < 8; ++i) { v[i] = bf2f(pk.u[i]); mx = fmaxf(mx, v[i]); }

    red[t] = mx; __syncthreads();
    for (int s = 128; s > 0; s >>= 1) {
        if (t < s) red[t] = fmaxf(red[t], red[t + s]);
        __syncthreads();
    }
    mx = red[0]; __syncthreads();

    float sum = 0.f;
#pragma unroll
    for (int i = 0; i < 8; ++i) { v[i] = __expf(v[i] - mx); sum += v[i]; }
    red[t] = sum; __syncthreads();
    for (int s = 128; s > 0; s >>= 1) {
        if (t < s) red[t] += red[t + s];
        __syncthreads();
    }
    const float inv = 1.0f / red[0];

#pragma unroll
    for (int i = 0; i < 8; ++i) pk.u[i] = f2bf(v[i] * inv);
    *(uint4*)(p + t * 8) = pk.q;
}

// ---------------------------------------------------------------------------
extern "C" void kernel_launch(void* const* d_in, const int* in_sizes, int n_in,
                              void* d_out, int out_size, void* d_ws, size_t ws_size,
                              hipStream_t stream) {
    const float* x  = (const float*)d_in[0];
    const float* Wq = (const float*)d_in[1];
    const float* bq = (const float*)d_in[2];
    const float* Wk = (const float*)d_in[3];
    const float* bk = (const float*)d_in[4];
    const float* Wv = (const float*)d_in[5];
    const float* bv = (const float*)d_in[6];

    const long long SZ_X   = (long long)ATT_B * ATT_S * ATT_D;  // 8,388,608
    const long long SZ_W   = (long long)ATT_D * ATT_H;          // 1,048,576
    const long long SZ_QKV = (long long)ATT_B * ATT_S * ATT_H;  // 8,388,608

    unsigned short* xb  = (unsigned short*)d_ws;
    unsigned short* wqt = xb  + SZ_X;
    unsigned short* wkt = wqt + SZ_W;
    unsigned short* wvt = wkt + SZ_W;
    unsigned short* Qb  = wvt + SZ_W;
    unsigned short* Kb  = Qb  + SZ_QKV;
    unsigned short* Vt  = Kb  + SZ_QKV;
    unsigned short* Sc  = Vt  + SZ_QKV;

    dim3 blk(256);

    // 1) x -> bf16
    convert_bf16_v4<<<(unsigned)((SZ_X / 4 + 255) / 256), blk, 0, stream>>>(x, xb, SZ_X / 4);

    // 2) W -> W^T bf16
    unsigned tgrid = (unsigned)((SZ_W + 255) / 256);
    transpose_bf16<<<tgrid, blk, 0, stream>>>(Wq, wqt, ATT_D, ATT_H);
    transpose_bf16<<<tgrid, blk, 0, stream>>>(Wk, wkt, ATT_D, ATT_H);
    transpose_bf16<<<tgrid, blk, 0, stream>>>(Wv, wvt, ATT_D, ATT_H);

    // 3) Q, K, V projections (V stored transposed for the P·V GEMM)
    dim3 gProj(ATT_H / BN, (ATT_B * ATT_S) / BM, 1);
    wmma_gemm<<<gProj, blk, 0, stream>>>(xb, wqt, bq, 1.0f, Qb,
                                         ATT_B * ATT_S, ATT_H, ATT_D,
                                         0, 0, 0, /*mode=*/0, ATT_S);
    wmma_gemm<<<gProj, blk, 0, stream>>>(xb, wkt, bk, 1.0f, Kb,
                                         ATT_B * ATT_S, ATT_H, ATT_D,
                                         0, 0, 0, /*mode=*/0, ATT_S);
    wmma_gemm<<<gProj, blk, 0, stream>>>(xb, wvt, bv, 1.0f, Vt,
                                         ATT_B * ATT_S, ATT_H, ATT_D,
                                         0, 0, 0, /*mode=*/1, ATT_S);

    // 4) scores = Q K^T / sqrt(H)   (batched over z)
    dim3 gSc(ATT_S / BN, ATT_S / BM, ATT_B);
    wmma_gemm<<<gSc, blk, 0, stream>>>(Qb, Kb, nullptr, 0.03125f, Sc,
                                       ATT_S, ATT_S, ATT_H,
                                       (long long)ATT_S * ATT_H,
                                       (long long)ATT_S * ATT_H,
                                       (long long)ATT_S * ATT_S, /*mode=*/0, ATT_S);

    // 5) softmax rows (in place, bf16)
    softmax_rows<<<ATT_B * ATT_S, blk, 0, stream>>>(Sc, ATT_S);

    // 6) O = P V  -> fp32 output
    dim3 gO(ATT_H / BN, ATT_S / BM, ATT_B);
    wmma_gemm<<<gO, blk, 0, stream>>>(Sc, Vt, nullptr, 1.0f, d_out,
                                      ATT_S, ATT_H, ATT_S,
                                      (long long)ATT_S * ATT_S,
                                      (long long)ATT_H * ATT_S,
                                      (long long)ATT_S * ATT_H, /*mode=*/2, ATT_S);
}
